// KANLinear_12068858101935
// MI455X (gfx1250) — compile-verified, matrix operand order checked by reading, not run on hardware
//
#include <hip/hip_runtime.h>
#include <hip/hip_bf16.h>

typedef __attribute__((ext_vector_type(16))) _Float16 v16h;
typedef __attribute__((ext_vector_type(8)))  _Float16 v8h;
typedef __attribute__((ext_vector_type(8)))  float    v8f;

#define N_ROWS   32768
#define F_DIM    512
#define U_DIM    512
#define K_TOT    3584          // 512 (base) + 512*6 (spline)
#define NKNOT    10

#define BM 128
#define BN 128
#define BK 32
#define PADK 40                // padded K-extent of LDS tiles (halves)

// -------- CDNA5 async global->LDS copy (ASYNCcnt-tracked, bypasses VGPRs) ----
__device__ __forceinline__ void async_ld_b128(unsigned lds_addr, const _Float16* g) {
  asm volatile("global_load_async_to_lds_b128 %0, %1, off"
               :: "v"(lds_addr), "v"(g) : "memory");
}
#define WAIT_ASYNCCNT(N) asm volatile("s_wait_asynccnt " #N ::: "memory")

// ---------------------------------------------------------------------------
// Kernel 1: LayerNorm + ReLU + cubic B-spline basis -> f16 activation matrix
// A[row] = [ relu(xn[0..511]) | basis(xn[f])[6] for f in 0..511 ]  (K_TOT halves)
// ---------------------------------------------------------------------------
__global__ __launch_bounds__(256) void kan_prep(
    const float* __restrict__ x, const float* __restrict__ gamma,
    const float* __restrict__ beta, const float* __restrict__ grid,
    _Float16* __restrict__ A)
{
  __shared__ float redS[256];
  __shared__ float redQ[256];
  const int row = blockIdx.x;
  const int t = threadIdx.x;
  const float x0 = x[(size_t)row * F_DIM + t];
  const float x1 = x[(size_t)row * F_DIM + t + 256];
  redS[t] = x0 + x1;
  redQ[t] = x0 * x0 + x1 * x1;
  __syncthreads();
  for (int s = 128; s > 0; s >>= 1) {
    if (t < s) { redS[t] += redS[t + s]; redQ[t] += redQ[t + s]; }
    __syncthreads();
  }
  const float mean = redS[0] * (1.0f / F_DIM);
  const float var  = redQ[0] * (1.0f / F_DIM) - mean * mean;
  const float rstd = rsqrtf(var + 1e-3f);

  _Float16* Arow = A + (size_t)row * K_TOT;

  #pragma unroll
  for (int e = 0; e < 2; ++e) {
    const int f = t + e * 256;
    const float xv = (e == 0) ? x0 : x1;
    const float xn = (xv - mean) * rstd * gamma[f] + beta[f];

    Arow[f] = (_Float16)fmaxf(xn, 0.0f);

    float tk[NKNOT];
    #pragma unroll
    for (int i = 0; i < NKNOT; ++i) tk[i] = grid[f * NKNOT + i];

    float b[9];
    #pragma unroll
    for (int i = 0; i < 9; ++i)
      b[i] = (xn >= tk[i] && xn < tk[i + 1]) ? 1.0f : 0.0f;
    #pragma unroll
    for (int i = 0; i < 8; ++i)   // k = 1
      b[i] = (xn - tk[i]) / (tk[i + 1] - tk[i]) * b[i]
           + (tk[i + 2] - xn) / (tk[i + 2] - tk[i + 1]) * b[i + 1];
    #pragma unroll
    for (int i = 0; i < 7; ++i)   // k = 2
      b[i] = (xn - tk[i]) / (tk[i + 2] - tk[i]) * b[i]
           + (tk[i + 3] - xn) / (tk[i + 3] - tk[i + 1]) * b[i + 1];
    #pragma unroll
    for (int i = 0; i < 6; ++i)   // k = 3
      b[i] = (xn - tk[i]) / (tk[i + 3] - tk[i]) * b[i]
           + (tk[i + 4] - xn) / (tk[i + 4] - tk[i + 1]) * b[i + 1];

    _Float16* Asp = Arow + F_DIM + f * 6;
    #pragma unroll
    for (int j = 0; j < 6; ++j) Asp[j] = (_Float16)b[j];
  }
}

// ---------------------------------------------------------------------------
// Kernel 2: pack [W_base ; W_spline] -> f16, K-TRANSPOSED: BT[n*K_TOT + k]
// (so GEMM can stage B tiles with contiguous-in-K async b128 copies, no
//  in-kernel transpose needed)
// ---------------------------------------------------------------------------
__global__ __launch_bounds__(256) void kan_pack_wT(
    const float* __restrict__ bw, const float* __restrict__ sw,
    _Float16* __restrict__ BT)
{
  const int idx = blockIdx.x * 256 + threadIdx.x;    // < K_TOT * U_DIM
  const int k = idx / U_DIM;
  const int n = idx - k * U_DIM;
  const float v = (k < F_DIM) ? bw[k * U_DIM + n]
                              : sw[(k - F_DIM) * U_DIM + n];
  BT[(size_t)n * K_TOT + k] = (_Float16)v;
}

// ---------------------------------------------------------------------------
// Kernel 3: C[N, U] = A[N, K] * BT[U, K]^T + bias
// f16 in / f32 WMMA acc; 256 threads (8 waves); block tile 128x128, BK=32;
// wave tile 32x64 -> 8 WMMAs per K-step; double-buffered LDS filled by
// async global->LDS b128 copies overlapped with compute.
// ---------------------------------------------------------------------------
__global__ __launch_bounds__(256) void kan_gemm(
    const _Float16* __restrict__ A, const _Float16* __restrict__ BT,
    const float* __restrict__ bias, float* __restrict__ out)
{
  __shared__ _Float16 As[2][BM * PADK];
  __shared__ _Float16 Bs[2][BN * PADK];

  const int t = threadIdx.x;
  const int lane = t & 31;
  const int wave = t >> 5;
  const int waveM = wave & 3;        // 4 M-wave groups (32 rows each)
  const int waveN = wave >> 2;       // 2 N-wave groups (64 cols each)
  const int half = lane >> 4;        // lanes 0-15 vs 16-31
  const int l16 = lane & 15;
  const int rowBase = blockIdx.y * BM;
  const int colBase = blockIdx.x * BN;

  // stage one 128x32 A tile + 128x32 B tile: 4 async b128 per thread
  auto stage = [&](int kt, int buf) {
    const int kBase = kt * BK;
    #pragma unroll
    for (int p = 0; p < 2; ++p) {
      const int c = t + p * 256;
      const int r = c >> 2;
      const int kc = (c & 3) * 8;
      async_ld_b128((unsigned)(size_t)&As[buf][r * PADK + kc],
                    A + (size_t)(rowBase + r) * K_TOT + kBase + kc);
    }
    #pragma unroll
    for (int p = 0; p < 2; ++p) {
      const int c = t + p * 256;
      const int n = c >> 2;
      const int kc = (c & 3) * 8;
      async_ld_b128((unsigned)(size_t)&Bs[buf][n * PADK + kc],
                    BT + (size_t)(colBase + n) * K_TOT + kBase + kc);
    }
  };

  v8f acc[2][4] = {};

  const int nkt = K_TOT / BK;        // 112 K-steps
  stage(0, 0);

  for (int kt = 0; kt < nkt; ++kt) {
    const int buf = kt & 1;
    if (kt + 1 < nkt) {
      stage(kt + 1, buf ^ 1);        // prefetch next tile into other buffer
      WAIT_ASYNCCNT(4);              // current tile's 4 copies done (in-order)
    } else {
      WAIT_ASYNCCNT(0);
    }
    __syncthreads();                 // all waves' staging visible

    // fragments (documented 16-bit layout: lane = M/N, K split at half*8, 16+half*8)
    v16h afrag[2], bfrag[4];
    #pragma unroll
    for (int mt = 0; mt < 2; ++mt) {
      const _Float16* ap = &As[buf][(waveM * 32 + mt * 16 + l16) * PADK];
      v8h lo = *(const v8h*)(ap + half * 8);
      v8h hi = *(const v8h*)(ap + 16 + half * 8);
      #pragma unroll
      for (int i = 0; i < 8; ++i) { afrag[mt][i] = lo[i]; afrag[mt][8 + i] = hi[i]; }
    }
    #pragma unroll
    for (int nt = 0; nt < 4; ++nt) {
      const _Float16* bp = &Bs[buf][(waveN * 64 + nt * 16 + l16) * PADK];
      v8h lo = *(const v8h*)(bp + half * 8);
      v8h hi = *(const v8h*)(bp + 16 + half * 8);
      #pragma unroll
      for (int i = 0; i < 8; ++i) { bfrag[nt][i] = lo[i]; bfrag[nt][8 + i] = hi[i]; }
    }

    #pragma unroll
    for (int mt = 0; mt < 2; ++mt)
      #pragma unroll
      for (int nt = 0; nt < 4; ++nt)
        acc[mt][nt] = __builtin_amdgcn_wmma_f32_16x16x32_f16(
            false, afrag[mt], false, bfrag[nt],
            (short)0, acc[mt][nt], false, false);

    __syncthreads();                 // reads done before buffer reuse
  }

  // epilogue: C/D layout -> M = half*8 + r, N = l16; add base bias
  #pragma unroll
  for (int mt = 0; mt < 2; ++mt) {
    #pragma unroll
    for (int nt = 0; nt < 4; ++nt) {
      const int gcol = colBase + waveN * 64 + nt * 16 + l16;
      const float bb = bias[gcol];
      #pragma unroll
      for (int r = 0; r < 8; ++r) {
        const int grow = rowBase + waveM * 32 + mt * 16 + half * 8 + r;
        out[(size_t)grow * U_DIM + gcol] = acc[mt][nt][r] + bb;
      }
    }
  }
}

// ---------------------------------------------------------------------------
extern "C" void kernel_launch(void* const* d_in, const int* in_sizes, int n_in,
                              void* d_out, int out_size, void* d_ws, size_t ws_size,
                              hipStream_t stream) {
  (void)in_sizes; (void)n_in; (void)out_size; (void)ws_size;
  const float* x     = (const float*)d_in[0];   // [N, F]
  const float* gamma = (const float*)d_in[1];   // [F]
  const float* beta  = (const float*)d_in[2];   // [F]
  const float* bw    = (const float*)d_in[3];   // [F, U]
  const float* bbias = (const float*)d_in[4];   // [U]
  const float* sw    = (const float*)d_in[5];   // [F*6, U]
  const float* grid  = (const float*)d_in[6];   // [1, F, 10]
  float* out = (float*)d_out;

  char* ws = (char*)d_ws;
  _Float16* Aact = (_Float16*)ws;                              // [N, K_TOT] f16
  const size_t aBytes = (size_t)N_ROWS * K_TOT * sizeof(_Float16);
  _Float16* BmatT = (_Float16*)(ws + aBytes);                  // [U, K_TOT] f16

  kan_prep<<<dim3(N_ROWS), dim3(256), 0, stream>>>(x, gamma, beta, grid, Aact);
  kan_pack_wT<<<dim3((K_TOT * U_DIM) / 256), dim3(256), 0, stream>>>(bw, sw, BmatT);
  kan_gemm<<<dim3(U_DIM / BN, N_ROWS / BM), dim3(256), 0, stream>>>(Aact, BmatT, bbias, out);
}
